// PureMambaBlock_54004918780242
// MI455X (gfx1250) — matrix operand order repaired and checked by reading, not compiled, not run on hardware
//
#include <hip/hip_runtime.h>
#include <hip/hip_bf16.h>
#include <stdint.h>

typedef unsigned short u16;
typedef __attribute__((ext_vector_type(16))) __bf16 v16bf;
typedef __attribute__((ext_vector_type(8)))  __bf16 v8bf;
typedef __attribute__((ext_vector_type(8)))  float  v8f;
typedef __attribute__((ext_vector_type(4)))  int    v4i;

#define BATCH   2
#define SEQLEN  2048
#define DMODEL  1024
#define DINNER  2048
#define DSTATE  16
#define DTRANK  64
#define MTOT    (BATCH*SEQLEN)   /* 4096 rows of (b,l) */
#define XPROJ_N (DTRANK + 2*DSTATE) /* 96 */

__device__ __forceinline__ u16 f2bf(float f){
  uint32_t u = __float_as_uint(f);
  u += 0x7fffu + ((u >> 16) & 1u);          // round-to-nearest-even
  return (u16)(u >> 16);
}
__device__ __forceinline__ float sigmoidf_(float x){ return 1.0f/(1.0f+__expf(-x)); }

// ---------------------------------------------------------------- fp32->bf16
__global__ void cvt_f32_bf16(const float* __restrict__ in, u16* __restrict__ out, int n){
  int i = blockIdx.x*blockDim.x + threadIdx.x;
  if (i < n) out[i] = f2bf(in[i]);
}

// ---------------------------------------------------------------- bf16 WMMA GEMM
// C[M,N] = A[M,K] * B[N,K]^T ; block tile 128x128, k-step 32, 8 waves/block,
// each wave does 2x4 = 8 v_wmma_f32_16x16x32_bf16 per k-step.
// Tile staging uses the CDNA5 async global->LDS path (ASYNCcnt) when the
// toolchain exposes the builtin, else classic load+ds_store.
#if __has_builtin(__builtin_amdgcn_global_load_async_to_lds_b128)
#define USE_ASYNC_LDS 1
typedef __attribute__((address_space(1))) v4i gv4i;   // global int4
typedef __attribute__((address_space(3))) v4i lv4i;   // LDS int4
#else
#define USE_ASYNC_LDS 0
#endif

__global__ __launch_bounds__(256)
void gemm_bf16_wmma(const u16* __restrict__ A, const u16* __restrict__ B,
                    float* __restrict__ C, int M, int N, int K,
                    int lda, int ldb, int ldc){
  __shared__ __align__(64) __bf16 As[128][32];
  __shared__ __align__(64) __bf16 Bs[128][32];
  const int tid  = threadIdx.x;
  const int lane = tid & 31;
  const int w    = tid >> 5;
  const int wm   = w & 3;          // 4 waves along M (32 rows each)
  const int wn   = w >> 2;         // 2 waves along N (64 cols each)
  const int mblock = blockIdx.y * 128;
  const int nblock = blockIdx.x * 128;

  v8f acc[2][4];
  #pragma unroll
  for (int mt=0; mt<2; ++mt)
    #pragma unroll
    for (int nt=0; nt<4; ++nt) acc[mt][nt] = (v8f)(0.0f);

  const int c0  = (lane >= 16) ? 8  : 0;   // A-fragment K offset is 2*c0 (contiguous k = 16h+s)
  const int bk  = (lane >= 16) ? 16 : 0;   // B-fragment K offset
  const int l15 = lane & 15;

  for (int k0 = 0; k0 < K; k0 += 32){
    // ---- cooperative stage of 128x32 A and B tiles (16B per thread chunk)
#if USE_ASYNC_LDS
    #pragma unroll
    for (int i=0;i<2;++i){
      int c   = tid + i*256;              // 0..511 chunks of 8 bf16
      int row = c >> 2;
      int col = (c & 3) * 8;
      __builtin_amdgcn_global_load_async_to_lds_b128(
          (gv4i*)(A + (size_t)(mblock+row)*lda + k0 + col),
          (lv4i*)&As[row][col], 0, 0);
      __builtin_amdgcn_global_load_async_to_lds_b128(
          (gv4i*)(B + (size_t)(nblock+row)*ldb + k0 + col),
          (lv4i*)&Bs[row][col], 0, 0);
    }
  #if __has_builtin(__builtin_amdgcn_s_wait_asynccnt)
    __builtin_amdgcn_s_wait_asynccnt(0);
  #else
    asm volatile("s_wait_asynccnt 0x0" ::: "memory");
  #endif
#else
    #pragma unroll
    for (int i=0;i<2;++i){
      int c   = tid + i*256;              // 0..511 chunks of 8 bf16
      int row = c >> 2;
      int col = (c & 3) * 8;
      uint4 va = *(const uint4*)(A + (size_t)(mblock+row)*lda + k0 + col);
      *(uint4*)&As[row][col] = va;
      uint4 vb = *(const uint4*)(B + (size_t)(nblock+row)*ldb + k0 + col);
      *(uint4*)&Bs[row][col] = vb;
    }
#endif
    __syncthreads();

    if (k0 + 32 < K){  // prefetch next k tile -> global_prefetch_b8
      __builtin_prefetch(A + (size_t)(mblock + (tid>>1))*lda + (k0+32), 0, 1);
      __builtin_prefetch(B + (size_t)(nblock + (tid>>1))*ldb + (k0+32), 0, 1);
    }

    // ---- build wave32 WMMA fragments from LDS (contiguous k = 16h+s in both A and B)
    v16bf afr[2], bfr[4];
    #pragma unroll
    for (int mt=0; mt<2; ++mt){
      const __bf16* p = &As[wm*32 + mt*16 + l15][0];
      v8bf a0 = *(const v8bf*)(p + 2*c0);      // K = 2*c0 .. 2*c0+7
      v8bf a1 = *(const v8bf*)(p + 2*c0 + 8);  // K = 2*c0+8 .. 2*c0+15
      afr[mt] = __builtin_shufflevector(a0, a1, 0,1,2,3,4,5,6,7,8,9,10,11,12,13,14,15);
    }
    #pragma unroll
    for (int nt=0; nt<4; ++nt)
      bfr[nt] = *(const v16bf*)&Bs[wn*64 + nt*16 + l15][bk]; // K = bk .. bk+15, N = l15

    #pragma unroll
    for (int mt=0; mt<2; ++mt)
      #pragma unroll
      for (int nt=0; nt<4; ++nt)
        acc[mt][nt] = __builtin_amdgcn_wmma_f32_16x16x32_bf16(
            false, afr[mt], false, bfr[nt], (short)0, acc[mt][nt], false, false);
    __syncthreads();
  }

  // ---- epilogue: C/D layout -> VGPR r holds M = r (+8 for lanes>=16), N = lane%16
  const int mrow0 = mblock + wm*32 + ((lane>=16)?8:0);
  const int ncol0 = nblock + wn*64 + l15;
  #pragma unroll
  for (int mt=0; mt<2; ++mt)
    #pragma unroll
    for (int nt=0; nt<4; ++nt)
      #pragma unroll
      for (int r=0; r<8; ++r)
        C[(size_t)(mrow0 + mt*16 + r)*ldc + ncol0 + nt*16] = acc[mt][nt][r];
}

// ---------------------------------------------------------------- causal depthwise conv + SiLU
// xz rows (m = b*L+l, 4096 cols): cols [0,2048) are x in (b,l,d) layout.
__global__ void conv_silu_kernel(const float* __restrict__ xz,
                                 const float* __restrict__ conv_w,
                                 const float* __restrict__ conv_b,
                                 float* __restrict__ xact){
  int idx = blockIdx.x*blockDim.x + threadIdx.x;   // over B*L*DINNER
  int d = idx & (DINNER-1);
  int l = (idx >> 11) & (SEQLEN-1);
  int b = idx >> 22;
  if (b >= BATCH) return;
  float acc = conv_b[d];
  #pragma unroll
  for (int j=0;j<4;++j){
    int ll = l - 3 + j;
    if (ll >= 0)
      acc += xz[((size_t)(b*SEQLEN + ll))*(2*DINNER) + d] * conv_w[d*4 + j];
  }
  acc = acc * sigmoidf_(acc);                      // SiLU
  xact[((size_t)(b*SEQLEN + l))*DINNER + d] = acc;
}

// ---------------------------------------------------------------- x_proj: (m,96) = xact(m,:) . W(96,2048)^T
__global__ __launch_bounds__(256)
void xproj_kernel(const float* __restrict__ xact,
                  const float* __restrict__ xproj_w,
                  float* __restrict__ xdbl){
  __shared__ float row[DINNER];
  const int m = blockIdx.x;
  for (int i = threadIdx.x; i < DINNER; i += 256)
    row[i] = xact[(size_t)m*DINNER + i];
  __syncthreads();
  int j = threadIdx.x;
  if (j < XPROJ_N){
    const float* wr = xproj_w + (size_t)j*DINNER;
    float acc = 0.f;
    for (int k=0;k<DINNER;++k) acc += row[k]*wr[k];
    xdbl[(size_t)m*XPROJ_N + j] = acc;
  }
}

// ---------------------------------------------------------------- dt_proj + softplus -> delta (b,l,d)
__global__ void dtproj_kernel(const float* __restrict__ xdbl,
                              const float* __restrict__ dtw,
                              const float* __restrict__ dtb,
                              float* __restrict__ delta){
  int idx = blockIdx.x*blockDim.x + threadIdx.x;   // over MTOT*DINNER
  int d = idx & (DINNER-1);
  int m = idx >> 11;
  if (m >= MTOT) return;
  const float* xr = xdbl + (size_t)m*XPROJ_N;
  const float* wr = dtw + (size_t)d*DTRANK;
  float acc = dtb[d];
  #pragma unroll 8
  for (int r=0;r<DTRANK;++r) acc += xr[r]*wr[r];
  float sp = (acc > 20.f) ? acc : __logf(1.0f + __expf(acc));  // softplus
  delta[(size_t)m*DINNER + d] = sp;
}

// ---------------------------------------------------------------- selective scan
// thread = (b, d, n); wave32 = 2 channels x 16 states; shfl_xor reduces over n.
// Fuses y = (scan_y + D*x) * silu(z) and emits bf16 (b,l,d) for out_proj.
__global__ __launch_bounds__(256)
void scan_kernel(const float* __restrict__ delta,
                 const float* __restrict__ xact,
                 const float* __restrict__ xdbl,
                 const float* __restrict__ xz,      // z at cols [2048,4096)
                 const float* __restrict__ A_log,
                 const float* __restrict__ Dp,
                 u16* __restrict__ ybf){
  const int lane = threadIdx.x & 31;
  const int w    = threadIdx.x >> 5;
  const int n    = lane & 15;
  const int d    = blockIdx.x*16 + w*2 + (lane >> 4);
  const int b    = blockIdx.y;
  const float Adn = -__expf(A_log[d*DSTATE + n]);
  const float Dd  = Dp[d];
  float state = 0.f;
  for (int l=0; l<SEQLEN; ++l){
    const int m = (b << 11) + l;
    float dl = delta[(size_t)m*DINNER + d];
    float xv = xact [(size_t)m*DINNER + d];
    float Bn = xdbl[(size_t)m*XPROJ_N + DTRANK + n];
    float Cn = xdbl[(size_t)m*XPROJ_N + DTRANK + DSTATE + n];
    state = __expf(dl*Adn)*state + dl*Bn*xv;
    float y = state*Cn;
    y += __shfl_xor(y, 1);
    y += __shfl_xor(y, 2);
    y += __shfl_xor(y, 4);
    y += __shfl_xor(y, 8);
    if (n == 0){
      float zv = xz[(size_t)m*(2*DINNER) + DINNER + d];
      float yf = (y + Dd*xv) * (zv * sigmoidf_(zv));
      ybf[(size_t)m*DINNER + d] = f2bf(yf);
    }
  }
}

// ---------------------------------------------------------------- launch
extern "C" void kernel_launch(void* const* d_in, const int* in_sizes, int n_in,
                              void* d_out, int out_size, void* d_ws, size_t ws_size,
                              hipStream_t stream) {
  const float* hidden = (const float*)d_in[0];
  const float* inw    = (const float*)d_in[1];
  const float* convw  = (const float*)d_in[2];
  const float* convb  = (const float*)d_in[3];
  const float* xprojw = (const float*)d_in[4];
  const float* dtw    = (const float*)d_in[5];
  const float* dtb    = (const float*)d_in[6];
  const float* alog   = (const float*)d_in[7];
  const float* Dp     = (const float*)d_in[8];
  const float* outw   = (const float*)d_in[9];
  float* out = (float*)d_out;

  char* ws = (char*)d_ws;
  size_t off = 0;
  auto alloc = [&](size_t bytes)->char*{
    char* p = ws + off; off = (off + bytes + 255) & ~(size_t)255; return p;
  };
  u16*   hidb  = (u16*)  alloc((size_t)MTOT*DMODEL*2);
  u16*   inwb  = (u16*)  alloc((size_t)2*DINNER*DMODEL*2);
  u16*   outwb = (u16*)  alloc((size_t)DMODEL*DINNER*2);
  float* xz    = (float*)alloc((size_t)MTOT*2*DINNER*4);
  float* xact  = (float*)alloc((size_t)MTOT*DINNER*4);
  float* xdbl  = (float*)alloc((size_t)MTOT*XPROJ_N*4);
  float* delta = (float*)alloc((size_t)MTOT*DINNER*4);
  u16*   ybf   = (u16*)  alloc((size_t)MTOT*DINNER*2);
  (void)ws_size; (void)in_sizes; (void)n_in; (void)out_size;

  // 1) fp32 -> bf16 packs
  {
    int n1 = MTOT*DMODEL;          // hidden
    int n2 = 2*DINNER*DMODEL;      // in_proj_w
    int n3 = DMODEL*DINNER;        // out_proj_w
    cvt_f32_bf16<<<(n1+255)/256, 256, 0, stream>>>(hidden, hidb, n1);
    cvt_f32_bf16<<<(n2+255)/256, 256, 0, stream>>>(inw,    inwb, n2);
    cvt_f32_bf16<<<(n3+255)/256, 256, 0, stream>>>(outw,  outwb, n3);
  }
  // 2) in_proj GEMM: xz[m,e] = hid[m,:] . inw[e,:]   (M=4096,N=4096,K=1024)
  gemm_bf16_wmma<<<dim3((2*DINNER)/128, MTOT/128), 256, 0, stream>>>(
      hidb, inwb, xz, MTOT, 2*DINNER, DMODEL, DMODEL, DMODEL, 2*DINNER);
  // 3) causal depthwise conv + SiLU
  {
    int total = BATCH*SEQLEN*DINNER;
    conv_silu_kernel<<<total/256, 256, 0, stream>>>(xz, convw, convb, xact);
  }
  // 4) x_proj
  xproj_kernel<<<MTOT, 256, 0, stream>>>(xact, xprojw, xdbl);
  // 5) dt_proj + softplus
  {
    int total = MTOT*DINNER;
    dtproj_kernel<<<total/256, 256, 0, stream>>>(xdbl, dtw, dtb, delta);
  }
  // 6) selective scan (fused gating) -> bf16 y in (b,l,d)
  scan_kernel<<<dim3(DINNER/16, BATCH), 256, 0, stream>>>(
      delta, xact, xdbl, xz, alog, Dp, ybf);
  // 7) out_proj GEMM: out[m,o] = y[m,:] . outw[o,:]   (M=4096,N=1024,K=2048)
  gemm_bf16_wmma<<<dim3(DMODEL/128, MTOT/128), 256, 0, stream>>>(
      ybf, outwb, out, MTOT, DMODEL, DINNER, DINNER, DINNER, DMODEL);
  // 8) residual (second tuple element) = input, appended flat after `out`
  (void)hipMemcpyAsync(out + (size_t)MTOT*DMODEL, hidden,
                       (size_t)MTOT*DMODEL*sizeof(float),
                       hipMemcpyDeviceToDevice, stream);
}
